// myWholeRGAT_13932873909018
// MI455X (gfx1250) — compile-verified, matrix operand order checked by reading, not run on hardware
//
#include <hip/hip_runtime.h>
#include <hip/hip_bf16.h>

// ---------------------------------------------------------------------------
// Fixed problem sizes from the reference
// ---------------------------------------------------------------------------
#define BG   4      // graphs
#define NG   512    // nodes per graph
#define NTOT 2048   // total nodes
#define FD   128    // feature dim
#define NL   3      // layers

typedef __attribute__((ext_vector_type(16))) _Float16 v16h;
typedef __attribute__((ext_vector_type(8)))  _Float16 v8h;
typedef __attribute__((ext_vector_type(8)))  float    v8f;

// Load a 16-bit WMMA A/B fragment where the 32 K-values for this lane live at
// [base .. base+7] and [base+16 .. base+23] (AMD table K order, two b128s).
__device__ __forceinline__ v16h load_frag(const _Float16* base) {
    v8h lo = *(const v8h*)(base);
    v8h hi = *(const v8h*)(base + 16);
    v16h r;
#pragma unroll
    for (int i = 0; i < 8; ++i) { r[i] = lo[i]; r[i + 8] = hi[i]; }
    return r;
}

__device__ __forceinline__ v8f wmma_f16(v16h a, v16h b, v8f c) {
    return __builtin_amdgcn_wmma_f32_16x16x32_f16(
        /*neg_a=*/false, a, /*neg_b=*/false, b,
        /*c_mod=*/(short)0, c, /*reuse_a=*/false, /*reuse_b=*/false);
}

// ---------------------------------------------------------------------------
// One-time weight preparation: W -> Wt f16 [L][256][128] (c-major, c=r*128+o),
// lin_W -> f16 copy [L][128][256]
// ---------------------------------------------------------------------------
__global__ void prep_weights_kernel(const float* __restrict__ W,
                                    const float* __restrict__ linW,
                                    _Float16* __restrict__ Wt,
                                    _Float16* __restrict__ lwh) {
    int idx = blockIdx.x * blockDim.x + threadIdx.x;   // 0 .. 3*256*128-1
    if (idx >= NL * 256 * FD) return;
    int l   = idx / (256 * FD);
    int rem = idx % (256 * FD);
    int c   = rem / FD;
    int k   = rem % FD;
    int r = c >> 7, o = c & 127;
    Wt[idx]  = (_Float16)W[((l * 2 + r) * FD + k) * FD + o];
    lwh[idx] = (_Float16)linW[idx];                    // same total element count
}

// ---------------------------------------------------------------------------
// x[(g*512+i)][f] = desc{0,1}[g][f][i]; also keep an f16 copy
// ---------------------------------------------------------------------------
__global__ void init_x_kernel(const float* __restrict__ d0,
                              const float* __restrict__ d1,
                              float* __restrict__ x,
                              _Float16* __restrict__ xh) {
    int idx = blockIdx.x * blockDim.x + threadIdx.x;   // NTOT*FD
    int n = idx >> 7, f = idx & 127;
    int g = n >> 9, i = n & 511;
    float v = (i < 256) ? d0[(g * FD + f) * 256 + i]
                        : d1[(g * FD + f) * 256 + (i - 256)];
    x[idx] = v;
    xh[idx] = (_Float16)v;
}

__global__ void zero_kernel(float* __restrict__ p, int n) {
    int idx = blockIdx.x * blockDim.x + threadIdx.x;
    if (idx < n) p[idx] = 0.f;
}

// ---------------------------------------------------------------------------
// proj: projT[c][node] = (x @ W[l])  (c = r*128+o), f16 output (transposed).
// Grid: 128 row tiles, 512 threads (16 waves = 16 column tiles of 16).
// ---------------------------------------------------------------------------
__global__ void proj_kernel(const _Float16* __restrict__ xh,   // [2048][128]
                            const _Float16* __restrict__ Wt,   // [256][128]
                            _Float16* __restrict__ projT) {    // [256][2048]
    int rt    = blockIdx.x;            // row tile
    int wave  = threadIdx.x >> 5;      // column tile 0..15
    int lane  = threadIdx.x & 31;
    int lhalf = lane >> 4, lidx = lane & 15, kbase = lhalf * 8;
    int row0  = rt * 16;
    int col   = wave * 16 + lidx;      // 0..255

    const _Float16* arow = xh + (row0 + lidx) * FD;
    const _Float16* bcol = Wt + col * FD;
    v8f acc = {};
#pragma unroll
    for (int kk = 0; kk < 4; ++kk) {
        int k0 = kk * 32;
        v16h a = load_frag(arow + k0 + kbase);
        v16h b = load_frag(bcol + k0 + kbase);
        acc = wmma_f16(a, b, acc);
    }
    // store transposed f16 tile: projT[col][row0 + 8*lhalf + i], contiguous
    v8h out;
#pragma unroll
    for (int i = 0; i < 8; ++i) out[i] = (_Float16)acc[i];
    *(v8h*)(projT + col * NTOT + row0 + 8 * lhalf) = out;
}

// ---------------------------------------------------------------------------
// Attention scalars from transposed proj: fully coalesced (lane = node).
// sq[n][r] = proj[n,r,:]·q ; sk[n][r] = proj[n,r,:]·k
// Grid: 8 blocks x 256 threads (one thread per node).
// ---------------------------------------------------------------------------
__global__ void scores_kernel(const _Float16* __restrict__ projT,  // [256][2048]
                              const float* __restrict__ qv,        // [128]
                              const float* __restrict__ kv,        // [128]
                              float* __restrict__ sq,              // [2048][2]
                              float* __restrict__ sk) {            // [2048][2]
    int n = blockIdx.x * blockDim.x + threadIdx.x;   // 0..2047
    float q0 = 0.f, k0 = 0.f, q1 = 0.f, k1 = 0.f;
#pragma unroll 4
    for (int o = 0; o < FD; ++o) {
        float qo = qv[o], ko = kv[o];                // scalar-uniform loads
        float p0 = (float)projT[o * NTOT + n];       // coalesced across lanes
        float p1 = (float)projT[(FD + o) * NTOT + n];
        q0 += p0 * qo; k0 += p0 * ko;
        q1 += p1 * qo; k1 += p1 * ko;
    }
    sq[n * 2 + 0] = q0; sk[n * 2 + 0] = k0;
    sq[n * 2 + 1] = q1; sk[n * 2 + 1] = k1;
}

// ---------------------------------------------------------------------------
// Dense attention: per (graph, 16-dst tile): scores over all 512 sources,
// row softmax, att(f16) x V(f16) via WMMA, m1 = relu(agg + conv_b) -> f16.
// Grid: (32, 4), 256 threads (8 waves -> 8 feature tiles of 16).
// ---------------------------------------------------------------------------
__global__ void att_kernel(const float* __restrict__ sq,
                           const float* __restrict__ sk,
                           const _Float16* __restrict__ projT,  // [256][2048]
                           const float* __restrict__ convb,     // [128]
                           _Float16* __restrict__ m1h) {        // [2048][128]
    __shared__ float    S[16][NG];
    __shared__ _Float16 A[16][NG];

    int g  = blockIdx.y;
    int dt = blockIdx.x;                 // 0..31
    int G  = (dt >= 16) ? 1 : 0;         // dst group
    int nodeBase = g * NG;
    int d0 = dt * 16;                    // dst base within graph
    int tid = threadIdx.x;

    // scores (with self-edge mask on intra relation)
    for (int idx = tid; idx < 16 * NG; idx += 256) {
        int di = idx >> 9;
        int j  = idx & 511;
        int jg = j >> 8;
        int rel = (jg == G) ? 0 : 1;
        float a;
        if (rel == 0 && j == d0 + di) {
            a = -1e30f;
        } else {
            float s = sq[(nodeBase + d0 + di) * 2 + rel]
                    + sk[(nodeBase + j) * 2 + rel];
            a = (s > 0.f) ? s : 0.2f * s;   // leaky_relu(0.2)
        }
        S[di][j] = a;
    }
    __syncthreads();

    // row softmax: 8 waves x 2 rows
    int wave = tid >> 5, lane = tid & 31;
#pragma unroll
    for (int rr = 0; rr < 2; ++rr) {
        int di = wave * 2 + rr;
        float mx = -1e30f;
        for (int j = lane; j < NG; j += 32) mx = fmaxf(mx, S[di][j]);
#pragma unroll
        for (int off = 1; off < 32; off <<= 1) mx = fmaxf(mx, __shfl_xor(mx, off, 32));
        float sum = 0.f;
        for (int j = lane; j < NG; j += 32) {
            float e = __expf(S[di][j] - mx);
            S[di][j] = e;
            sum += e;
        }
#pragma unroll
        for (int off = 1; off < 32; off <<= 1) sum += __shfl_xor(sum, off, 32);
        float inv = 1.f / (sum + 1e-16f);
        for (int j = lane; j < NG; j += 32) A[di][j] = (_Float16)(S[di][j] * inv);
    }
    __syncthreads();

    // att x V : M=16, N=128 (wave -> 16-col tile), K=512
    int lhalf = lane >> 4, lidx = lane & 15, kbase = lhalf * 8;
    int ocol = wave * 16 + lidx;         // output feature 0..127
    v8f acc = {};
#pragma unroll
    for (int kk = 0; kk < 16; ++kk) {
        int k0 = kk * 32;
        int jg = (k0 >= 256) ? 1 : 0;    // source group for this K block
        int rel = (jg == G) ? 0 : 1;
        v16h a = load_frag(&A[lidx][k0 + kbase]);
        const _Float16* bp = projT + (rel * FD + ocol) * NTOT + nodeBase + k0 + kbase;
        v16h b = load_frag(bp);
        acc = wmma_f16(a, b, acc);
    }
    float cb = convb[ocol];
#pragma unroll
    for (int i = 0; i < 8; ++i) {
        int node = nodeBase + d0 + i + 8 * lhalf;
        float v = acc[i] + cb;
        m1h[node * FD + ocol] = (_Float16)fmaxf(v, 0.f);
    }
}

// ---------------------------------------------------------------------------
// m2 = concat(x, m1) @ lin_W^T ; also per-feature sum / sumsq via atomics.
// (lin_b cancels exactly under BatchNorm, so it is omitted.)
// Grid: 128 row tiles, 256 threads (8 waves = 8 feature tiles).
// ---------------------------------------------------------------------------
__global__ void lin_kernel(const _Float16* __restrict__ xh,    // [2048][128]
                           const _Float16* __restrict__ m1h,   // [2048][128]
                           const _Float16* __restrict__ lw,    // [128][256] f16
                           float* __restrict__ m2,             // [2048][128]
                           float* __restrict__ sumBuf,         // [128]
                           float* __restrict__ sumSqBuf) {     // [128]
    int rt    = blockIdx.x;
    int wave  = threadIdx.x >> 5;
    int lane  = threadIdx.x & 31;
    int lhalf = lane >> 4, lidx = lane & 15, kbase = lhalf * 8;
    int row0  = rt * 16;
    int f     = wave * 16 + lidx;

    const _Float16* bcol = lw + f * 256;     // B[k=c][n=f] = lin_W[f][c]
    v8f acc = {};
#pragma unroll
    for (int kk = 0; kk < 8; ++kk) {
        int k0 = kk * 32;
        const _Float16* arow = (k0 < FD)
            ? (xh  + (row0 + lidx) * FD + k0)
            : (m1h + (row0 + lidx) * FD + (k0 - FD));
        v16h a = load_frag(arow + kbase);
        v16h b = load_frag(bcol + k0 + kbase);
        acc = wmma_f16(a, b, acc);
    }
    float s = 0.f, s2 = 0.f;
#pragma unroll
    for (int i = 0; i < 8; ++i) {
        int n = row0 + i + 8 * lhalf;
        m2[n * FD + f] = acc[i];
        s  += acc[i];
        s2 += acc[i] * acc[i];
    }
    s  += __shfl_xor(s, 16, 32);
    s2 += __shfl_xor(s2, 16, 32);
    if (lhalf == 0) {
        atomicAdd(&sumBuf[f], s);
        atomicAdd(&sumSqBuf[f], s2);
    }
}

__global__ void bn_stats_kernel(const float* __restrict__ sum,
                                const float* __restrict__ sumsq,
                                const float* __restrict__ gamma,
                                const float* __restrict__ beta,
                                float* __restrict__ aC,
                                float* __restrict__ bC) {
    int f = threadIdx.x;                 // 128 threads
    const float invN = 1.f / (float)NTOT;
    float mu  = sum[f] * invN;
    float var = sumsq[f] * invN - mu * mu;
    float inv = rsqrtf(var + 1e-5f);
    float a = gamma[f] * inv;
    aC[f] = a;
    bC[f] = beta[f] - mu * a;
}

__global__ void update_kernel(const float* __restrict__ m2,
                              const float* __restrict__ aC,
                              const float* __restrict__ bC,
                              float* __restrict__ x,
                              _Float16* __restrict__ xh) {
    int idx = blockIdx.x * blockDim.x + threadIdx.x;   // NTOT*FD
    int f = idx & 127;
    float v = x[idx] + m2[idx] * aC[f] + bC[f];
    x[idx] = v;
    xh[idx] = (_Float16)v;
}

__global__ void out_kernel(const float* __restrict__ x, float* __restrict__ out) {
    int idx = blockIdx.x * blockDim.x + threadIdx.x;   // 2 * 4*128*256
    int half = idx / (BG * FD * 256);
    int rem  = idx % (BG * FD * 256);
    int g = rem >> 15;                 // / (128*256)
    int f = (rem >> 8) & 127;
    int i = rem & 255;
    int n = g * NG + half * 256 + i;
    out[idx] = x[n * FD + f];
}

// ---------------------------------------------------------------------------
// Host side
// ---------------------------------------------------------------------------
extern "C" void kernel_launch(void* const* d_in, const int* in_sizes, int n_in,
                              void* d_out, int out_size, void* d_ws, size_t ws_size,
                              hipStream_t stream) {
    (void)in_sizes; (void)n_in; (void)out_size; (void)ws_size;

    const float* desc0 = (const float*)d_in[0];
    const float* desc1 = (const float*)d_in[1];
    const float* W     = (const float*)d_in[2];   // [3][2][128][128]
    const float* qAll  = (const float*)d_in[3];   // [3][128][1]
    const float* kAll  = (const float*)d_in[4];   // [3][128][1]
    const float* convb = (const float*)d_in[5];   // [3][128]
    const float* linW  = (const float*)d_in[6];   // [3][128][256]
    const float* gamma = (const float*)d_in[8];   // [3][128]
    const float* beta  = (const float*)d_in[9];   // [3][128]

    char* ws = (char*)d_ws;
    float*    x     = (float*)   (ws + 0);            // 1,048,576 B
    float*    m2    = (float*)   (ws + 1048576);      // 1,048,576 B
    _Float16* xh    = (_Float16*)(ws + 2097152);      //   524,288 B
    _Float16* m1h   = (_Float16*)(ws + 2621440);      //   524,288 B
    _Float16* projT = (_Float16*)(ws + 3145728);      // 1,048,576 B
    float*    sq    = (float*)   (ws + 4194304);      //    16,384 B
    float*    sk    = (float*)   (ws + 4210688);      //    16,384 B
    float*    sumB  = (float*)   (ws + 4227072);      //       512 B
    float*    sumSq = (float*)   (ws + 4227584);      //       512 B
    float*    aC    = (float*)   (ws + 4228096);      //       512 B
    float*    bC    = (float*)   (ws + 4228608);      //       512 B
    _Float16* Wt    = (_Float16*)(ws + 4229120);      //   196,608 B
    _Float16* lwh   = (_Float16*)(ws + 4425728);      //   196,608 B

    // one-time (re-run every call for determinism; cheap)
    prep_weights_kernel<<<(NL * 256 * FD + 255) / 256, 256, 0, stream>>>(W, linW, Wt, lwh);
    init_x_kernel<<<(NTOT * FD) / 256, 256, 0, stream>>>(desc0, desc1, x, xh);

    for (int l = 0; l < NL; ++l) {
        // zero BN accumulators (sumB, sumSq contiguous: 256 floats)
        zero_kernel<<<1, 256, 0, stream>>>(sumB, 256);

        proj_kernel<<<NTOT / 16, 512, 0, stream>>>(xh, Wt + l * 256 * FD, projT);

        scores_kernel<<<NTOT / 256, 256, 0, stream>>>(
            projT, qAll + l * FD, kAll + l * FD, sq, sk);

        att_kernel<<<dim3(NG / 16, BG), 256, 0, stream>>>(
            sq, sk, projT, convb + l * FD, m1h);

        lin_kernel<<<NTOT / 16, 256, 0, stream>>>(
            xh, m1h, lwh + l * FD * 256, m2, sumB, sumSq);

        bn_stats_kernel<<<1, FD, 0, stream>>>(
            sumB, sumSq, gamma + l * FD, beta + l * FD, aC, bC);

        update_kernel<<<(NTOT * FD) / 256, 256, 0, stream>>>(m2, aC, bC, x, xh);
    }

    out_kernel<<<(2 * BG * FD * 256) / 256, 256, 0, stream>>>(x, (float*)d_out);
}
